// Model_67010079752826
// MI455X (gfx1250) — matrix operand order repaired and checked by reading, not compiled, not run on hardware
//
#include <hip/hip_runtime.h>
#include <hip/hip_bf16.h>
#include <math.h>

// ---- problem constants (from reference) ----
#define BB      2048
#define SS      64
#define AA      14
#define EE      16
#define DD      224          // A*E
#define FF      227          // D+3
#define FPAD    256          // padded feature row stride (multiple of 32, 16B aligned)
#define HH      10
#define NHEADS  4
#define DHD     56           // D/HEADS
#define NLAYERS 2
#define DECH    64
#define GAP     4
#define NBS     (BB*SS)      // 131072

typedef __attribute__((ext_vector_type(16))) _Float16 v16h;
typedef __attribute__((ext_vector_type(8)))  float    v8f;
typedef __attribute__((ext_vector_type(8)))  _Float16 h8v;
typedef __attribute__((ext_vector_type(4)))  _Float16 h4v;
typedef __attribute__((ext_vector_type(4)))  float    f4v;

// ============================================================================
// Weight pack: W (K x N, f32 row-major) -> f16 B-fragments in the exact CDNA5
// wave32 layout.  One thread per (ntile16, kblock, lane); writes 16 halfs.
// Zero-pads K up to Kpad; N must be a multiple of 16.
// Packed index: ((tn*nkb + kbi)*32 + lane)*16
// ============================================================================
__global__ void pack_w_kernel(const float* __restrict__ W, _Float16* __restrict__ P,
                              int N, int K, int Kpad)
{
    int i = blockIdx.x * blockDim.x + threadIdx.x;
    const int nkb = Kpad >> 5;
    const int Nt  = N >> 4;
    if (i >= Nt * nkb * 32) return;
    int lane = i & 31;
    int kbi  = (i >> 5) % nkb;
    int tn   = (i >> 5) / nkb;
    int half = lane >> 4;
    int n    = (tn << 4) + (lane & 15);
    _Float16* dst = P + (size_t)i * 16;
#pragma unroll
    for (int j = 0; j < 8; ++j) {
        int kk = (kbi << 5) + 8*half + ((j >= 4) ? 16 : 0) + 2*(j & 3);
        float b0 = (kk     < K) ? W[(size_t)kk * N + n]       : 0.f;
        float b1 = (kk + 1 < K) ? W[(size_t)(kk + 1) * N + n] : 0.f;
        dst[2*j]   = (_Float16)b0;
        dst[2*j+1] = (_Float16)b1;
    }
}

// vectorized f32 -> f16 conversion (n multiple of 4)
__global__ void f32_to_f16_kernel(const float* __restrict__ src,
                                  _Float16* __restrict__ dst, long n)
{
    long i = ((long)blockIdx.x * blockDim.x + threadIdx.x) * 4;
    if (i >= n) return;
    f4v a = *(const f4v*)(src + i);
    h4v h;
#pragma unroll
    for (int j = 0; j < 4; ++j) h[j] = (_Float16)a[j];
    *(h4v*)(dst + i) = h;
}

// ============================================================================
// Guard-free WMMA GEMM, 16x32 C tile per wave (2 accumulators share 1 A frag):
//   C[M,N] = act( A[M,K] @ Wpacked + bias [+ C] )
//  * K multiple of 32, N multiple of 32, M multiple of 16
//  * A row-major stride lda (lda in {224,256} -> vector loads 16B aligned)
//  * B pre-packed in wave32 fragment order (2x b128 loads per WMMA)
// 8 waves per 256-thread block.
// ============================================================================
template<bool A_HALF, bool C_HALF, bool RELU, bool RESID>
__global__ void gemm_wmma_kernel(const void* __restrict__ Ap, int lda,
                                 const _Float16* __restrict__ Bp,
                                 const float* __restrict__ bias,
                                 void* __restrict__ Cp,
                                 int M, int N, int K)
{
    const int lane = threadIdx.x & 31;
    const int wave = threadIdx.x >> 5;
    const int Nt2 = N >> 5;                    // 32-wide tiles
    const int nkb = K >> 5;
    const long total = (long)(M >> 4) * Nt2;
    const long tile  = (long)blockIdx.x * 8 + wave;
    if (tile >= total) return;                 // wave-uniform: EXEC stays all-1s

    const int tm = (int)(tile / Nt2);
    const int tn = (int)(tile % Nt2);
    const int m0 = tm << 4, n0 = tn << 5;
    const int half = lane >> 4;
    const int lq   = lane & 15;
    const int am  = m0 + lq;                   // A row held by this lane
    const int cn0 = n0 + lq;                   // C column (tile 0)
    const int cn1 = cn0 + 16;                  // C column (tile 1)

    const float*    Af = (const float*)Ap;
    const _Float16* Ah = (const _Float16*)Ap;
    const _Float16* Bl0 = Bp + ((size_t)(2*tn)     * nkb * 32 + lane) * 16;
    const _Float16* Bl1 = Bp + ((size_t)(2*tn + 1) * nkb * 32 + lane) * 16;

    v8f acc0 = {0.f,0.f,0.f,0.f,0.f,0.f,0.f,0.f};
    v8f acc1 = {0.f,0.f,0.f,0.f,0.f,0.f,0.f,0.f};

    for (int kbi = 0; kbi < nkb; ++kbi) {
        const int kb = kbi << 5;
        v16h af;
        if (A_HALF) {
            const _Float16* ar = Ah + (size_t)am * lda + kb + 8*half;
            h8v r0 = *(const h8v*)ar;          // K = kb+8h .. +7
            h8v r1 = *(const h8v*)(ar + 16);   // K = kb+16+8h .. +7
#pragma unroll
            for (int j = 0; j < 8; ++j) { af[j] = r0[j]; af[8+j] = r1[j]; }
        } else {
            const float* ar = Af + (size_t)am * lda + kb + 8*half;
            f4v a0 = *(const f4v*)ar;
            f4v a1 = *(const f4v*)(ar + 4);
            f4v a2 = *(const f4v*)(ar + 16);
            f4v a3 = *(const f4v*)(ar + 20);
#pragma unroll
            for (int j = 0; j < 4; ++j) {
                af[j]    = (_Float16)a0[j];
                af[4+j]  = (_Float16)a1[j];
                af[8+j]  = (_Float16)a2[j];
                af[12+j] = (_Float16)a3[j];
            }
        }
        const _Float16* br0 = Bl0 + (size_t)kbi * 32 * 16;
        const _Float16* br1 = Bl1 + (size_t)kbi * 32 * 16;
        h8v b00 = *(const h8v*)br0;
        h8v b01 = *(const h8v*)(br0 + 8);
        h8v b10 = *(const h8v*)br1;
        h8v b11 = *(const h8v*)(br1 + 8);
        v16h bf0, bf1;
#pragma unroll
        for (int j = 0; j < 8; ++j) {
            bf0[j] = b00[j]; bf0[8+j] = b01[j];
            bf1[j] = b10[j]; bf1[8+j] = b11[j];
        }
        acc0 = __builtin_amdgcn_wmma_f32_16x16x32_f16(
            false, af, false, bf0, (short)0, acc0, false, false);
        acc1 = __builtin_amdgcn_wmma_f32_16x16x32_f16(
            false, af, false, bf1, (short)0, acc1, false, false);
    }

    const float bv0 = bias ? bias[cn0] : 0.f;
    const float bv1 = bias ? bias[cn1] : 0.f;
    float*    Cf = (float*)Cp;
    _Float16* Ch = (_Float16*)Cp;
#pragma unroll
    for (int r = 0; r < 8; ++r) {
        const int row = m0 + (half ? 8 : 0) + r;
        const long o0 = (long)row * N + cn0;
        const long o1 = (long)row * N + cn1;
        float v0 = acc0[r] + bv0;
        float v1 = acc1[r] + bv1;
        if (RESID) {
            v0 += C_HALF ? (float)Ch[o0] : Cf[o0];
            v1 += C_HALF ? (float)Ch[o1] : Cf[o1];
        }
        if (RELU) { v0 = v0 > 0.f ? v0 : 0.f; v1 = v1 > 0.f ? v1 : 0.f; }
        if (C_HALF) { Ch[o0] = (_Float16)v0; Ch[o1] = (_Float16)v1; }
        else        { Cf[o0] = v0;           Cf[o1] = v1; }
    }
}

// ============================================================================
// init: cur_masks = masks copy, zero loss accumulators
// ============================================================================
__global__ void init_kernel(const float* __restrict__ masks,
                            float* __restrict__ cm, float* __restrict__ accum)
{
    int i = blockIdx.x * blockDim.x + threadIdx.x;
    if (i < NBS) cm[i] = masks[i];
    if (i < 8)   accum[i] = 0.f;
}

// zero the padding columns [FF, FPAD) of feat -- required for K=256 GEMMs
__global__ void pad_feat_kernel(float* __restrict__ feat)
{
    int i = blockIdx.x * blockDim.x + threadIdx.x;
    const int PADW = FPAD - FF;                 // 29
    if (i >= NBS * PADW) return;
    int r = i / PADW, f = FF + i % PADW;
    feat[(long)r * FPAD + f] = 0.f;
}

// ============================================================================
// encode: feat[:,:, :D] = x = ais[...,a]*enc_W[a,e] + enc_b[a,e]
// ============================================================================
__global__ void encode_full_kernel(const float* __restrict__ ais,
                                   const float* __restrict__ encW,
                                   const float* __restrict__ encb,
                                   float* __restrict__ feat,
                                   float* __restrict__ x)
{
    long i = (long)blockIdx.x * blockDim.x + threadIdx.x;
    if (i >= (long)NBS * DD) return;
    int  d = (int)(i % DD);
    long r = i / DD;
    int  a = d >> 4;
    float vv = ais[r*AA + a] * encW[d] + encb[d];
    feat[r*FPAD + d] = vv;
    x[i] = vv;
}

// ============================================================================
// Attention: one block per (b, head), 64 threads (one per query row).
// q/k/v are f16 (B*S, D), head h = columns [h*56, h*56+56).
// Writes softmax(qk^T/sqrt(dh)) @ v back over the q buffer (LDS-staged, safe).
// ============================================================================
__global__ void attention_kernel(const _Float16* __restrict__ q,
                                 const _Float16* __restrict__ k,
                                 const _Float16* __restrict__ v,
                                 _Float16* __restrict__ o)
{
    __shared__ _Float16 sq[SS*DHD];
    __shared__ _Float16 sk[SS*DHD];
    __shared__ _Float16 sv[SS*DHD];
    const int bh = blockIdx.x;
    const int b  = bh / NHEADS;
    const int h  = bh % NHEADS;
    const long base = (long)b * SS * DD + h * DHD;

    for (int i = threadIdx.x; i < SS*DHD; i += blockDim.x) {
        int s = i / DHD, d = i % DHD;
        long g = base + (long)s * DD + d;
        sq[i] = q[g]; sk[i] = k[g]; sv[i] = v[g];
    }
    __syncthreads();

    const int s = threadIdx.x;
    const float scale = 0.13363062095621219f;   // 1/sqrt(56)
    float sc[SS];
    float mx = -3.0e38f;
    for (int t = 0; t < SS; ++t) {
        float acc = 0.f;
        for (int d = 0; d < DHD; ++d)
            acc += (float)sq[s*DHD + d] * (float)sk[t*DHD + d];
        acc *= scale;
        sc[t] = acc;
        mx = fmaxf(mx, acc);
    }
    float sum = 0.f;
    for (int t = 0; t < SS; ++t) { sc[t] = __expf(sc[t] - mx); sum += sc[t]; }
    float inv = 1.f / sum;
    for (int d = 0; d < DHD; ++d) {
        float acc = 0.f;
        for (int t = 0; t < SS; ++t) acc += sc[t] * (float)sv[t*DHD + d];
        o[base + (long)s * DD + d] = (_Float16)(acc * inv);
    }
}

// ============================================================================
// forces: out[r, ooff..ooff+2] = x[r,:224] @ W_force + b_force  (N=3, VALU)
// ============================================================================
__global__ void forces_kernel(const float* __restrict__ x, int rows,
                              const float* __restrict__ Wf, const float* __restrict__ bf,
                              float* __restrict__ out, int ostride, int ooff)
{
    int r = blockIdx.x * blockDim.x + threadIdx.x;
    if (r >= rows) return;
    float a0 = bf[0], a1 = bf[1], a2 = bf[2];
    const float* xr = x + (long)r * DD;
    for (int kk = 0; kk < DD; ++kk) {
        float xv = xr[kk];
        a0 += xv * Wf[kk*3 + 0];
        a1 += xv * Wf[kk*3 + 1];
        a2 += xv * Wf[kk*3 + 2];
    }
    float* op = out + (long)r * ostride + ooff;
    op[0] = a0; op[1] = a1; op[2] = a2;
}

// ============================================================================
// BRITS first-row: one thread per batch b.
//  * idx = first s with cur_mask==0 (else 0)
//  * delta scan + BRITS recurrence up to idx (d_t@W_gamma = delta*colsum(W_gamma))
//  * emit first_imputed[b,:] (row stride FPAD, padding zeroed)
// ============================================================================
__global__ void brits_first_kernel(const float* __restrict__ ais,
                                   const float* __restrict__ cmask,
                                   const float* __restrict__ feat,
                                   const float* __restrict__ Wg, const float* __restrict__ bg,
                                   const float* __restrict__ Wh, const float* __restrict__ Wx,
                                   const float* __restrict__ bh,
                                   const float* __restrict__ Wout, const float* __restrict__ bout,
                                   float* __restrict__ first_imp, int* __restrict__ idxbuf)
{
    int b = blockIdx.x * blockDim.x + threadIdx.x;
    if (b >= BB) return;

    int idx = 0; bool found = false;
    for (int s = 0; s < SS; ++s)
        if (!found && cmask[b*SS + s] == 0.f) { idx = s; found = true; }

    float sg[HH];
#pragma unroll
    for (int h = 0; h < HH; ++h) sg[h] = 0.f;
    for (int f = 0; f < FF; ++f)
#pragma unroll
        for (int h = 0; h < HH; ++h) sg[h] += Wg[f*HH + h];

    float hh[HH];
#pragma unroll
    for (int h = 0; h < HH; ++h) hh[h] = 0.f;

    float dlt = 0.f, tprev = 0.f, mprev = 0.f;
    for (int s = 0; s <= idx; ++s) {
        float t = ais[((long)b*SS + s) * AA];             // times = ais[...,0]
        if (s > 0) dlt = (dlt + (t - tprev)) * (1.f - mprev);
        float m = cmask[b*SS + s];

#pragma unroll
        for (int h = 0; h < HH; ++h) {
            float z = dlt * sg[h] + bg[h];
            z = z > 0.f ? z : 0.f;
            hh[h] *= __expf(-z);
        }
        float accH[HH];
#pragma unroll
        for (int h = 0; h < HH; ++h) accH[h] = bh[h];
        for (int h2 = 0; h2 < HH; ++h2) {
            float hv = hh[h2];
#pragma unroll
            for (int h = 0; h < HH; ++h) accH[h] += hv * Wh[h2*HH + h];
        }
        const float* Xr = feat + ((long)b*SS + s) * FPAD;
        bool last = (s == idx);
        for (int f = 0; f < FF; ++f) {
            float xh = bout[f];
#pragma unroll
            for (int h = 0; h < HH; ++h) xh += hh[h] * Wout[h*FF + f];
            float xc = m * Xr[f] + (1.f - m) * xh;
            if (last) first_imp[(long)b*FPAD + f] = xc;
#pragma unroll
            for (int h = 0; h < HH; ++h) accH[h] += xc * Wx[f*HH + h];
        }
#pragma unroll
        for (int h = 0; h < HH; ++h) hh[h] = tanhf(accH[h]);

        tprev = t; mprev = m;
    }
    for (int f = FF; f < FPAD; ++f) first_imp[(long)b*FPAD + f] = 0.f;  // K padding
    idxbuf[b] = idx;
}

// decode stage 2 for the single imputed row: (B,64) -> (B,14)
__global__ void small_decode2_kernel(const float* __restrict__ hid,
                                     const float* __restrict__ W2, const float* __restrict__ b2,
                                     float* __restrict__ out)
{
    int r = blockIdx.x * blockDim.x + threadIdx.x;
    if (r >= BB*AA) return;
    int b = r / AA, a = r % AA;
    float acc = b2[a];
    const float* hb = hid + (long)b * DECH;
    for (int kk = 0; kk < DECH; ++kk) acc += hb[kk] * W2[kk*AA + a];
    out[r] = acc;
}

// re-encode the decoded row: (B,14) -> (B,224)
__global__ void encode_rows_kernel(const float* __restrict__ dec,
                                   const float* __restrict__ encW,
                                   const float* __restrict__ encb,
                                   float* __restrict__ out)
{
    int i = blockIdx.x * blockDim.x + threadIdx.x;
    if (i >= BB*DD) return;
    int d = i % DD, b = i / DD;
    int a = d >> 4;
    out[i] = dec[b*AA + a] * encW[d] + encb[d];
}

// scatter first_features into feat[b, idx[b], :] and set cur_mask=1
__global__ void scatter_kernel(const float* __restrict__ fx,
                               const float* __restrict__ ffr,
                               const int* __restrict__ idxb,
                               float* __restrict__ feat, float* __restrict__ cm)
{
    int i = blockIdx.x * blockDim.x + threadIdx.x;
    if (i >= BB*FF) return;
    int b = i / FF, f = i % FF;
    int row = idxb[b];
    float vv = (f < DD) ? fx[(long)b*DD + f] : ffr[b*3 + (f - DD)];
    feat[((long)b*SS + row) * FPAD + f] = vv;
    if (f == 0) cm[b*SS + row] = 1.f;
}

// fused decode stage-2 + masked loss partial sums
__global__ void decode2_loss_kernel(const float* __restrict__ hid,
                                    const float* __restrict__ W2, const float* __restrict__ b2,
                                    const float* __restrict__ ais,
                                    const float* __restrict__ masks,
                                    float* __restrict__ accum)
{
    int r = blockIdx.x * 256 + threadIdx.x;
    float sa = 0.f, sn = 0.f, sd = 0.f;
    if (r < NBS) {
        const float* hb = hid + (long)r * DECH;
        float neg = 1.f - masks[r];
        sn = neg;
        for (int a = 0; a < AA; ++a) {
            float o = b2[a];
            for (int kk = 0; kk < DECH; ++kk) o += hb[kk] * W2[kk*AA + a];
            float xv = ais[(long)r*AA + a];
            sa += fabsf(o - xv) * neg;
            sd += fabsf(xv) + fabsf(o);
        }
    }
    __shared__ float red[3*256];
    red[threadIdx.x] = sa; red[256 + threadIdx.x] = sn; red[512 + threadIdx.x] = sd;
    __syncthreads();
    for (int st = 128; st > 0; st >>= 1) {
        if ((int)threadIdx.x < st) {
            red[threadIdx.x]       += red[threadIdx.x + st];
            red[256 + threadIdx.x] += red[256 + threadIdx.x + st];
            red[512 + threadIdx.x] += red[512 + threadIdx.x + st];
        }
        __syncthreads();
    }
    if (threadIdx.x == 0) {
        atomicAdd(&accum[0], red[0]);
        atomicAdd(&accum[1], red[256]);
        atomicAdd(&accum[2], red[512]);
    }
}

__global__ void finalize_kernel(const float* __restrict__ accum, float* __restrict__ out)
{
    if (threadIdx.x == 0) {
        out[0] = accum[0] / (accum[1] + 1e-9f);
        out[1] = accum[0] / (accum[2] * 0.5f + 1e-9f);
    }
}

// ============================================================================
static inline int gemm_blocks(int M, int N)
{
    long t = (long)(M >> 4) * (N >> 5);   // 16x32 tiles per wave, 8 waves/block
    return (int)((t + 7) / 8);
}

extern "C" void kernel_launch(void* const* d_in, const int* in_sizes, int n_in,
                              void* d_out, int out_size, void* d_ws, size_t ws_size,
                              hipStream_t stream)
{
    (void)in_sizes; (void)n_in; (void)out_size; (void)ws_size;
    const float* ais   = (const float*)d_in[0];
    const float* masks = (const float*)d_in[1];
    // d_in[2] = gap_size (always 4, compiled in)
    const float* encW  = (const float*)d_in[3];
    const float* encb  = (const float*)d_in[4];
    const float* Wq    = (const float*)d_in[5];
    const float* Wk    = (const float*)d_in[6];
    const float* Wv    = (const float*)d_in[7];
    const float* Wo    = (const float*)d_in[8];
    const float* Wf    = (const float*)d_in[9];
    const float* bfv   = (const float*)d_in[10];
    const float* Wg    = (const float*)d_in[11];
    const float* bg    = (const float*)d_in[12];
    const float* Wh    = (const float*)d_in[13];
    const float* Wx    = (const float*)d_in[14];
    const float* bh    = (const float*)d_in[15];
    const float* Wout  = (const float*)d_in[16];
    const float* bout  = (const float*)d_in[17];
    const float* dW1   = (const float*)d_in[18];
    const float* db1   = (const float*)d_in[19];
    const float* dW2   = (const float*)d_in[20];
    const float* db2   = (const float*)d_in[21];

    char* ws = (char*)d_ws;
    size_t cur = 0;
    auto alloc = [&](size_t bytes) -> char* {
        char* p = ws + cur;
        cur += (bytes + 255) & ~(size_t)255;
        return p;
    };
    float*    feat = (float*)   alloc((size_t)NBS * FPAD * 4);  // padded features
    float*    x    = (float*)   alloc((size_t)NBS * DD * 4);    // AFA working buf (f32)
    _Float16* xh   = (_Float16*)alloc((size_t)NBS * DD * 2);    // f16 mirror of x
    _Float16* q    = (_Float16*)alloc((size_t)NBS * DD * 2);    // f16 Q (reused as O)
    _Float16* kk   = (_Float16*)alloc((size_t)NBS * DD * 2);    // f16 K
    _Float16* vv   = (_Float16*)alloc((size_t)NBS * DD * 2);    // f16 V
    float*    hid  = (float*)   alloc((size_t)NBS * DECH * 4);  // decoder hidden
    float*    cm   = (float*)   alloc((size_t)NBS * 4);         // cur_masks
    int*      idxb = (int*)     alloc((size_t)BB * 4);
    float*    fimp = (float*)   alloc((size_t)BB * FPAD * 4);
    float*    fdh  = (float*)   alloc((size_t)BB * DECH * 4);
    float*    fdec = (float*)   alloc((size_t)BB * AA * 4);
    float*    fx   = (float*)   alloc((size_t)BB * DD * 4);
    float*    tmpv = (float*)   alloc((size_t)BB * DD * 4);
    float*    ffr  = (float*)   alloc((size_t)BB * 3 * 4);
    float*    acc  = (float*)   alloc(64);
    // packed f16 weights (B-fragment order)
    const size_t szDD = (size_t)(DD/16) * (DD/32) * 32 * 16;    // 50176 halfs
    const size_t szW1 = (size_t)(DECH/16) * (FPAD/32) * 32 * 16;
    _Float16* pQ[NLAYERS], *pK[NLAYERS], *pV[NLAYERS], *pO[NLAYERS];
    for (int l = 0; l < NLAYERS; ++l) {
        pQ[l] = (_Float16*)alloc(szDD * 2);
        pK[l] = (_Float16*)alloc(szDD * 2);
        pV[l] = (_Float16*)alloc(szDD * 2);
        pO[l] = (_Float16*)alloc(szDD * 2);
    }
    _Float16* pW1 = (_Float16*)alloc(szW1 * 2);

    // ---- weight packing (once per call; tiny, L2-resident) ----
    {
        int thrDD = (DD/16) * (DD/32) * 32;     // 3136
        int thrW1 = (DECH/16) * (FPAD/32) * 32; // 1024
        for (int l = 0; l < NLAYERS; ++l) {
            pack_w_kernel<<<(thrDD+255)/256,256,0,stream>>>(Wq + (size_t)l*DD*DD, pQ[l], DD, DD, DD);
            pack_w_kernel<<<(thrDD+255)/256,256,0,stream>>>(Wk + (size_t)l*DD*DD, pK[l], DD, DD, DD);
            pack_w_kernel<<<(thrDD+255)/256,256,0,stream>>>(Wv + (size_t)l*DD*DD, pV[l], DD, DD, DD);
            pack_w_kernel<<<(thrDD+255)/256,256,0,stream>>>(Wo + (size_t)l*DD*DD, pO[l], DD, DD, DD);
        }
        pack_w_kernel<<<(thrW1+255)/256,256,0,stream>>>(dW1, pW1, DECH, FF, FPAD);
    }

    init_kernel<<<(NBS + 255) / 256, 256, 0, stream>>>(masks, cm, acc);
    {
        long n = (long)NBS * DD;
        encode_full_kernel<<<(int)((n + 255) / 256), 256, 0, stream>>>(ais, encW, encb, feat, x);
    }
    pad_feat_kernel<<<(NBS*(FPAD-FF) + 255) / 256, 256, 0, stream>>>(feat);

    const long NXD = (long)NBS * DD;
    const int  cvtBlocks = (int)((NXD/4 + 255) / 256);
    f32_to_f16_kernel<<<cvtBlocks, 256, 0, stream>>>(x, xh, NXD);

    // ---- full AFA (2 layers) ----
    for (int l = 0; l < NLAYERS; ++l) {
        gemm_wmma_kernel<true, true, false, false>
            <<<gemm_blocks(NBS, DD), 256, 0, stream>>>(xh, DD, pQ[l], nullptr, q,  NBS, DD, DD);
        gemm_wmma_kernel<true, true, false, false>
            <<<gemm_blocks(NBS, DD), 256, 0, stream>>>(xh, DD, pK[l], nullptr, kk, NBS, DD, DD);
        gemm_wmma_kernel<true, true, false, false>
            <<<gemm_blocks(NBS, DD), 256, 0, stream>>>(xh, DD, pV[l], nullptr, vv, NBS, DD, DD);
        attention_kernel<<<BB * NHEADS, SS, 0, stream>>>(q, kk, vv, q);   // o overwrites q
        gemm_wmma_kernel<true, false, false, true>
            <<<gemm_blocks(NBS, DD), 256, 0, stream>>>(q, DD, pO[l], nullptr, x, NBS, DD, DD);
        if (l + 1 < NLAYERS)
            f32_to_f16_kernel<<<cvtBlocks, 256, 0, stream>>>(x, xh, NXD);
    }
    forces_kernel<<<(NBS + 255) / 256, 256, 0, stream>>>(x, NBS, Wf, bfv, feat, FPAD, DD);

    // ---- 4 imputation iterations ----
    for (int it = 0; it < GAP; ++it) {
        brits_first_kernel<<<(BB + 255) / 256, 256, 0, stream>>>(
            ais, cm, feat, Wg, bg, Wh, Wx, bh, Wout, bout, fimp, idxb);
        gemm_wmma_kernel<false, false, true, false>
            <<<gemm_blocks(BB, DECH), 256, 0, stream>>>(fimp, FPAD, pW1, db1, fdh, BB, DECH, FPAD);
        small_decode2_kernel<<<(BB*AA + 255) / 256, 256, 0, stream>>>(fdh, dW2, db2, fdec);
        encode_rows_kernel<<<(BB*DD + 255) / 256, 256, 0, stream>>>(fdec, encW, encb, fx);
        // s==1 attention: softmax over a singleton == 1  =>  o == v
        for (int l = 0; l < NLAYERS; ++l) {
            gemm_wmma_kernel<false, false, false, false>
                <<<gemm_blocks(BB, DD), 256, 0, stream>>>(fx, DD, pV[l], nullptr, tmpv, BB, DD, DD);
            gemm_wmma_kernel<false, false, false, true>
                <<<gemm_blocks(BB, DD), 256, 0, stream>>>(tmpv, DD, pO[l], nullptr, fx, BB, DD, DD);
        }
        forces_kernel<<<(BB + 255) / 256, 256, 0, stream>>>(fx, BB, Wf, bfv, ffr, 3, 0);
        scatter_kernel<<<(BB*FF + 255) / 256, 256, 0, stream>>>(fx, ffr, idxb, feat, cm);
    }

    // ---- final decode + loss ----
    gemm_wmma_kernel<false, false, true, false>
        <<<gemm_blocks(NBS, DECH), 256, 0, stream>>>(feat, FPAD, pW1, db1, hid, NBS, DECH, FPAD);
    decode2_loss_kernel<<<(NBS + 255) / 256, 256, 0, stream>>>(hid, dW2, db2, ais, masks, acc);
    finalize_kernel<<<1, 32, 0, stream>>>(acc, (float*)d_out);
}
